// HeadRoutingBackboneV5_47553877902014
// MI455X (gfx1250) — matrix operand order repaired and checked by reading, not compiled, not run on hardware
//
#include <hip/hip_runtime.h>

#define B_   2048
#define V_   3
#define K_   32
#define DIN_ 256
#define H_   8
#define DH_  64
#define REP_ 512
#define OBS_ 128
#define HID_ 256

typedef _Float16 h16;
typedef __attribute__((ext_vector_type(16))) _Float16 v16h;
typedef __attribute__((ext_vector_type(8)))  _Float16 v8h;
typedef __attribute__((ext_vector_type(8)))  float    v8f;

// ---------------------------------------------------------------------------
// helpers
// ---------------------------------------------------------------------------
__device__ __forceinline__ v8f wmma16(v16h a, v16h b, v8f c) {
  // D = A(16x32 f16) * B(32x16 f16) + C(16x16 f32)
  return __builtin_amdgcn_wmma_f32_16x16x32_f16(false, a, false, b, (short)0, c,
                                                false, false);
}

// A-fragment (16-bit A 16x32): lane L holds row M=L%16; elements 0..7 are
// K = kbase..kbase+7, elements 8..15 are K = kbase+16..kbase+23, where the
// caller passes p = &row[kstep*32 + 8*(L/16)].  Two aligned 16B chunks.
__device__ __forceinline__ v16h ld_afrag(const h16* p) {
  v8h lo = *(const v8h*)p;
  v8h hi = *(const v8h*)(p + 16);
  v16h r;
#pragma unroll
  for (int i = 0; i < 8; ++i) { r[i] = lo[i]; r[8 + i] = hi[i]; }
  return r;
}

__device__ __forceinline__ float wred_sum(float x) {
#pragma unroll
  for (int o = 16; o > 0; o >>= 1) x += __shfl_xor(x, o, 32);
  return x;
}
__device__ __forceinline__ float wred_max(float x) {
#pragma unroll
  for (int o = 16; o > 0; o >>= 1) x = fmaxf(x, __shfl_xor(x, o, 32));
  return x;
}
__device__ __forceinline__ float gelu_exact(float x) {
  return 0.5f * x * (1.0f + erff(x * 0.70710678118654752f));
}

// ---------------------------------------------------------------------------
// Weight packing: row-major f32 [Kd x Nd] -> f16 B-operand fragments.
// Packed index: ((nt*(Kd/32)+ks)*32 + lane)*16 + e, with
//   K = ks*32 + 16*(e/8) + 8*(lane/16) + e%8 ,  N = nt*16 + lane%16
// so the GEMM does one aligned v16h load per (nt,ks,lane).
// ---------------------------------------------------------------------------
__global__ void k_pack(const float* __restrict__ W, h16* __restrict__ out,
                       int Kd, int Nd, int nmats) {
  int per = Kd * Nd;
  long long total = (long long)per * nmats;
  int KS = Kd >> 5;
  for (long long idx = blockIdx.x * (long long)blockDim.x + threadIdx.x;
       idx < total; idx += (long long)gridDim.x * blockDim.x) {
    int m = (int)(idx / per);
    int r = (int)(idx - (long long)m * per);
    int e = r & 15;
    int lane = (r >> 4) & 31;
    int blk = r >> 9;
    int ks = blk % KS;
    int nt = blk / KS;
    int kk = ks * 32 + ((e >> 3) << 4) + ((lane >> 4) << 3) + (e & 7);
    int nn = nt * 16 + (lane & 15);
    out[idx] = (h16)W[(size_t)m * per + (size_t)kk * Nd + nn];
  }
}

// ---------------------------------------------------------------------------
// Encoder: 32 rows x 512 cols per workgroup (2 M-tiles per wave -> each B
// fragment feeds 2 WMMAs).  D = LN(X @ W + b) * g + beta, LN done from
// registers with cross-lane reductions (no full output staging in LDS).
// mode 0: anchor rows (ri = b), mode 1: neighbor rows (ri = b*32 + k).
// ---------------------------------------------------------------------------
__global__ __launch_bounds__(256) void k_enc(
    const float* __restrict__ X, const h16* __restrict__ Wpk,
    const float* __restrict__ bias, const float* __restrict__ gamma,
    const float* __restrict__ betap, float* __restrict__ outF,
    h16* __restrict__ outH, int mode) {
  __shared__ h16 Ah[32 * 264];  // f16 A tile, padded stride
  __shared__ float psum[32][8];
  __shared__ float psq[32][8];
  __shared__ float mu_s[32];
  __shared__ float rs_s[32];

  const int v = blockIdx.y;
  const int tile = blockIdx.x;
  const int tid = threadIdx.x;

  { // stage A tile (32 x 256 f32 -> f16 LDS); thread t: row t/8, 32 elems
    int row = tid >> 3, seg = tid & 7;
    int ri = tile * 32 + row;
    size_t xoff;
    if (mode == 0) {
      xoff = ((size_t)ri * V_ + v) * DIN_;
    } else {
      int b = ri >> 5, k = ri & 31;
      xoff = (((size_t)b * V_ + v) * K_ + k) * DIN_;
    }
    const float* xp = X + xoff + seg * 32;
    h16* ap = Ah + row * 264 + seg * 32;
#pragma unroll
    for (int j = 0; j < 32; ++j) ap[j] = (h16)xp[j];
  }
  __syncthreads();

  const int wave = tid >> 5, lane = tid & 31;
  const h16* Bbase = Wpk + (size_t)v * (DIN_ * REP_);
  const h16* arow0 = Ah + (lane & 15) * 264;
  const h16* arow1 = Ah + (16 + (lane & 15)) * 264;

  v8f acc[2][4];
#pragma unroll
  for (int mt = 0; mt < 2; ++mt)
#pragma unroll
    for (int j = 0; j < 4; ++j)
#pragma unroll
      for (int r = 0; r < 8; ++r) acc[mt][j][r] = 0.0f;

#pragma unroll
  for (int ks = 0; ks < 8; ++ks) {
    int ko = ks * 32 + ((lane >> 4) << 3);
    v16h a0 = ld_afrag(arow0 + ko);
    v16h a1 = ld_afrag(arow1 + ko);
#pragma unroll
    for (int j = 0; j < 4; ++j) {
      int nt = wave * 4 + j;
      const h16* bp = Bbase + (((size_t)(nt * 8 + ks) * 32 + lane) << 4);
      if (ks < 7) __builtin_prefetch(bp + 512, 0, 1);  // next K-step block
      v16h bfr = *(const v16h*)bp;
      acc[0][j] = wmma16(a0, bfr, acc[0][j]);
      acc[1][j] = wmma16(a1, bfr, acc[1][j]);
    }
  }

  // bias add (D layout: lane L col N = nt*16 + L%16, VGPR r row M=r+8*(L/16))
#pragma unroll
  for (int j = 0; j < 4; ++j) {
    int col = (wave * 4 + j) * 16 + (lane & 15);
    float bcol = bias[v * REP_ + col];
#pragma unroll
    for (int mt = 0; mt < 2; ++mt)
#pragma unroll
      for (int r = 0; r < 8; ++r) acc[mt][j][r] += bcol;
  }

  // per-row partial sums: reduce over the 16 lanes holding each row
#pragma unroll
  for (int mt = 0; mt < 2; ++mt) {
#pragma unroll
    for (int r = 0; r < 8; ++r) {
      float s = 0.f, q = 0.f;
#pragma unroll
      for (int j = 0; j < 4; ++j) {
        float x = acc[mt][j][r];
        s += x; q += x * x;
      }
#pragma unroll
      for (int o = 1; o < 16; o <<= 1) {
        s += __shfl_xor(s, o, 32);
        q += __shfl_xor(q, o, 32);
      }
      if ((lane & 15) == 0) {
        int row = mt * 16 + r + ((lane >> 4) << 3);
        psum[row][wave] = s;
        psq[row][wave] = q;
      }
    }
  }
  __syncthreads();
  if (tid < 32) {
    float s = 0.f, q = 0.f;
#pragma unroll
    for (int w = 0; w < 8; ++w) { s += psum[tid][w]; q += psq[tid][w]; }
    float mu = s * (1.f / 512.f);
    float var = q * (1.f / 512.f) - mu * mu;
    mu_s[tid] = mu;
    rs_s[tid] = rsqrtf(var + 1e-5f);
  }
  __syncthreads();

  // apply LN from registers and store
#pragma unroll
  for (int j = 0; j < 4; ++j) {
    int col = (wave * 4 + j) * 16 + (lane & 15);
    float gcol = gamma[v * REP_ + col];
    float bcol = betap[v * REP_ + col];
#pragma unroll
    for (int mt = 0; mt < 2; ++mt) {
#pragma unroll
      for (int r = 0; r < 8; ++r) {
        int row = mt * 16 + r + ((lane >> 4) << 3);
        float y = (acc[mt][j][r] - mu_s[row]) * rs_s[row] * gcol + bcol;
        int ri = tile * 32 + row;
        size_t obase;
        if (mode == 0) {
          obase = ((size_t)ri * V_ + v) * REP_;
        } else {
          int b = ri >> 5, k = ri & 31;
          obase = (((size_t)b * V_ + v) * K_ + k) * REP_;
        }
        outH[obase + col] = (h16)y;
        if (outF) outF[obase + col] = y;
      }
    }
  }
}

// ---------------------------------------------------------------------------
// q projection: per (v,h), q[B,64] = haH[:, v, h*64:+64] @ Wq[v,h].
// 32 rows per workgroup, wave = N-tile, B fragment reused across 2 M-tiles.
// ---------------------------------------------------------------------------
__global__ __launch_bounds__(128) void k_qgemm(const h16* __restrict__ haH,
                                               const h16* __restrict__ Wpk,
                                               float* __restrict__ qF) {
  int tile = blockIdx.x;
  int vh = blockIdx.y;
  int v = vh / H_, h = vh % H_;
  int tid = threadIdx.x, wave = tid >> 5, lane = tid & 31;  // wave == n-tile
  v8f acc[2];
#pragma unroll
  for (int mt = 0; mt < 2; ++mt)
#pragma unroll
    for (int r = 0; r < 8; ++r) acc[mt][r] = 0.f;
#pragma unroll
  for (int ks = 0; ks < 2; ++ks) {
    int row0 = tile * 32 + (lane & 15);
    const h16* ap0 = haH + ((size_t)row0 * V_ + v) * REP_ + h * DH_ + ks * 32 +
                     ((lane >> 4) << 3);
    const h16* ap1 = ap0 + (size_t)16 * V_ * REP_;
    v16h a0 = ld_afrag(ap0);
    v16h a1 = ld_afrag(ap1);
    const h16* bp = Wpk + (size_t)vh * 4096 +
                    (((size_t)(wave * 2 + ks) * 32 + lane) << 4);
    v16h b = *(const v16h*)bp;
    acc[0] = wmma16(a0, b, acc[0]);
    acc[1] = wmma16(a1, b, acc[1]);
  }
  int col = wave * 16 + (lane & 15);
#pragma unroll
  for (int mt = 0; mt < 2; ++mt)
#pragma unroll
    for (int r = 0; r < 8; ++r) {
      int b = tile * 32 + mt * 16 + r + ((lane >> 4) << 3);
      qF[(((size_t)b * V_ + v) * H_ + h) * DH_ + col] = acc[mt][r];
    }
}

// ---------------------------------------------------------------------------
// Fused per-(b,v) attention: k/v projections (WMMA, A fragments loaded once
// and kept live across both passes; B fragments reused across M-tiles),
// geometric softmax, weighted sum, Wo projection + residual.
// One wave per head (8 waves).
// ---------------------------------------------------------------------------
__global__ __launch_bounds__(256) void k_attn(
    const h16* __restrict__ hnH, const float* __restrict__ qF,
    const h16* __restrict__ Wkpk, const h16* __restrict__ Wvpk,
    const float* __restrict__ Wo, const float* __restrict__ gls,
    const float* __restrict__ ew, const float* __restrict__ haF,
    float* __restrict__ vtF) {
  __shared__ h16   kv[H_][K_][66];  // k then (overwritten) v, padded
  __shared__ float qb[H_][DH_];
  __shared__ float pb[H_][K_];
  __shared__ float ob[H_][DH_];

  int b = blockIdx.x, v = blockIdx.y;
  int tid = threadIdx.x, h = tid >> 5, lane = tid & 31;
  size_t bv = (size_t)b * V_ + v;

  qb[h][2 * lane]     = qF[(bv * H_ + h) * DH_ + 2 * lane];
  qb[h][2 * lane + 1] = qF[(bv * H_ + h) * DH_ + 2 * lane + 1];

  const h16* Arow0 = hnH + (bv * K_) * REP_ + h * DH_;
  const h16* Bk = Wkpk + (size_t)(v * H_ + h) * 4096;
  const h16* Bv = Wvpk + (size_t)(v * H_ + h) * 4096;

  // A fragments: [mt][ks], loaded once, reused for both k and v passes
  v16h afr[2][2];
#pragma unroll
  for (int mt = 0; mt < 2; ++mt) {
    const h16* ap =
        Arow0 + (size_t)(mt * 16 + (lane & 15)) * REP_ + ((lane >> 4) << 3);
    afr[mt][0] = ld_afrag(ap);
    afr[mt][1] = ld_afrag(ap + 32);
  }

  // ---- k = hn @ Wk  (32x64x64) ----
#pragma unroll
  for (int nt = 0; nt < 4; ++nt) {
    v16h b0 = *(const v16h*)(Bk + (((size_t)(nt * 2 + 0) * 32 + lane) << 4));
    v16h b1 = *(const v16h*)(Bk + (((size_t)(nt * 2 + 1) * 32 + lane) << 4));
    int col = nt * 16 + (lane & 15);
#pragma unroll
    for (int mt = 0; mt < 2; ++mt) {
      v8f acc;
#pragma unroll
      for (int r = 0; r < 8; ++r) acc[r] = 0.f;
      acc = wmma16(afr[mt][0], b0, acc);
      acc = wmma16(afr[mt][1], b1, acc);
#pragma unroll
      for (int r = 0; r < 8; ++r)
        kv[h][mt * 16 + r + ((lane >> 4) << 3)][col] = (h16)acc[r];
    }
  }
  __syncthreads();

  // ---- scores + softmax (lane = neighbor) ----
  float geo;
  { float x = gls[v * H_ + h]; geo = (x > 20.f) ? x : log1pf(expf(x)); }
  float s = 0.f;
#pragma unroll
  for (int d = 0; d < DH_; ++d) s += qb[h][d] * (float)kv[h][lane][d];
  s = s * 0.125f + geo * logf(ew[bv * K_ + lane] + 1e-6f);
  float mx = wred_max(s);
  float p = expf(s - mx);
  float den = wred_sum(p);
  pb[h][lane] = p / den;

  // ---- v = hn @ Wv (overwrite kv; same wave owns kv[h]) ----
#pragma unroll
  for (int nt = 0; nt < 4; ++nt) {
    v16h b0 = *(const v16h*)(Bv + (((size_t)(nt * 2 + 0) * 32 + lane) << 4));
    v16h b1 = *(const v16h*)(Bv + (((size_t)(nt * 2 + 1) * 32 + lane) << 4));
    int col = nt * 16 + (lane & 15);
#pragma unroll
    for (int mt = 0; mt < 2; ++mt) {
      v8f acc;
#pragma unroll
      for (int r = 0; r < 8; ++r) acc[r] = 0.f;
      acc = wmma16(afr[mt][0], b0, acc);
      acc = wmma16(afr[mt][1], b1, acc);
#pragma unroll
      for (int r = 0; r < 8; ++r)
        kv[h][mt * 16 + r + ((lane >> 4) << 3)][col] = (h16)acc[r];
    }
  }
  __syncthreads();

  // ---- weighted sum over neighbors (lane owns 2 dims) ----
  int d0 = 2 * lane, d1 = d0 + 1;
  float o0 = 0.f, o1 = 0.f;
#pragma unroll
  for (int k = 0; k < K_; ++k) {
    float pk = pb[h][k];
    o0 += pk * (float)kv[h][k][d0];
    o1 += pk * (float)kv[h][k][d1];
  }
  ob[h][d0] = o0;
  ob[h][d1] = o1;
  __syncthreads();

  // ---- Wo projection + residual ----
  const float* wo = Wo + (size_t)(v * H_ + h) * DH_ * DH_;
  float r0 = 0.f, r1 = 0.f;
#pragma unroll 8
  for (int din = 0; din < DH_; ++din) {
    float x = ob[h][din];
    r0 += x * wo[din * DH_ + d0];
    r1 += x * wo[din * DH_ + d1];
  }
  size_t hb = bv * REP_ + h * DH_;
  r0 += haF[hb + d0];
  r1 += haF[hb + d1];
  size_t ot = (((size_t)b * H_ + h) * V_ + v) * DH_;
  vtF[ot + d0] = r0;
  vtF[ot + d1] = r1;
}

// ---------------------------------------------------------------------------
// Router MLP (exact GELU) -> pi / beta / gate
// ---------------------------------------------------------------------------
__global__ __launch_bounds__(256) void k_router(
    const float* __restrict__ gin, const float* __restrict__ rW1,
    const float* __restrict__ rb1, const float* __restrict__ rW2,
    const float* __restrict__ rb2, const float* __restrict__ vW,
    const float* __restrict__ vb, const float* __restrict__ mW,
    const float* __restrict__ mb, const float* __restrict__ gW,
    const float* __restrict__ gbp, float* __restrict__ piF,
    float* __restrict__ betaF, float* __restrict__ gateF) {
  __shared__ float gs[OBS_];
  __shared__ float h1[HID_];
  __shared__ float h2[HID_];
  __shared__ float lv[24], lm[8], lg[8];
  int b = blockIdx.x, t = threadIdx.x;
  if (t < OBS_) gs[t] = gin[(size_t)b * OBS_ + t];
  __syncthreads();
  {
    float a = rb1[t];
#pragma unroll 8
    for (int i = 0; i < OBS_; ++i) a += gs[i] * rW1[i * HID_ + t];
    h1[t] = gelu_exact(a);
  }
  __syncthreads();
  {
    float a = rb2[t];
#pragma unroll 8
    for (int i = 0; i < HID_; ++i) a += h1[i] * rW2[i * HID_ + t];
    h2[t] = gelu_exact(a);
  }
  __syncthreads();
  if (t < 24) {
    float s = vb[t];
    for (int i = 0; i < HID_; ++i) s += h2[i] * vW[i * 24 + t];
    lv[t] = s;
  } else if (t < 32) {
    int j = t - 24;
    float s = mb[j];
    for (int i = 0; i < HID_; ++i) s += h2[i] * mW[i * 8 + j];
    lm[j] = s;
  } else if (t < 40) {
    int j = t - 32;
    float s = gbp[j];
    for (int i = 0; i < HID_; ++i) s += h2[i] * gW[i * 8 + j];
    lg[j] = s;
  }
  __syncthreads();
  if (t < 8) {
    int h = t;
    float x0 = lv[h * 3], x1 = lv[h * 3 + 1], x2 = lv[h * 3 + 2];
    float mx = fmaxf(x0, fmaxf(x1, x2));
    float e0 = expf(x0 - mx), e1 = expf(x1 - mx), e2 = expf(x2 - mx);
    float d = e0 + e1 + e2;
    piF[b * 24 + h * 3 + 0] = e0 / d;
    piF[b * 24 + h * 3 + 1] = e1 / d;
    piF[b * 24 + h * 3 + 2] = e2 / d;
    betaF[b * 8 + h] = 1.f / (1.f + expf(-lm[h]));
    float gm = lg[0];
    for (int j = 1; j < 8; ++j) gm = fmaxf(gm, lg[j]);
    float gsum = 0.f;
    for (int j = 0; j < 8; ++j) gsum += expf(lg[j] - gm);
    gateF[b * 8 + h] = expf(lg[h] - gm) / gsum;
  }
}

// ---------------------------------------------------------------------------
// Final: mixture + cross-view attention with log-pi prior + gating.
// One wave per head; lane owns 2 dims.
// ---------------------------------------------------------------------------
__global__ __launch_bounds__(256) void k_final(
    const float* __restrict__ vtF, const float* __restrict__ piF,
    const float* __restrict__ betaF, const float* __restrict__ gateF,
    const float* __restrict__ cWq, const float* __restrict__ cWk,
    const float* __restrict__ cWv, const float* __restrict__ cWo,
    float* __restrict__ out) {
  __shared__ float tb[H_][V_][DH_];
  __shared__ float mbuf[H_][DH_];
  __shared__ float cbuf[H_][DH_];
  int b = blockIdx.x, tid = threadIdx.x, h = tid >> 5, lane = tid & 31;

  const float* vrow = vtF + ((size_t)b * H_ + h) * V_ * DH_;
  for (int i = lane; i < V_ * DH_; i += 32) tb[h][i / DH_][i % DH_] = vrow[i];
  float p0 = piF[b * 24 + h * 3 + 0];
  float p1 = piF[b * 24 + h * 3 + 1];
  float p2 = piF[b * 24 + h * 3 + 2];
  __syncthreads();

  int d0 = 2 * lane, d1 = d0 + 1;
  float m0 = p0 * tb[h][0][d0] + p1 * tb[h][1][d0] + p2 * tb[h][2][d0];
  float m1 = p0 * tb[h][0][d1] + p1 * tb[h][1][d1] + p2 * tb[h][2][d1];
  mbuf[h][d0] = m0;
  mbuf[h][d1] = m1;
  __syncthreads();

  const float* wq = cWq + (size_t)h * DH_ * DH_;
  const float* wk = cWk + (size_t)h * DH_ * DH_;
  const float* wv = cWv + (size_t)h * DH_ * DH_;
  float qc0 = 0.f, qc1 = 0.f;
  float kc[V_][2] = {{0, 0}, {0, 0}, {0, 0}};
  float vc[V_][2] = {{0, 0}, {0, 0}, {0, 0}};
#pragma unroll 4
  for (int din = 0; din < DH_; ++din) {
    float mm = mbuf[h][din];
    float wq0 = wq[din * DH_ + d0], wq1 = wq[din * DH_ + d1];
    qc0 += mm * wq0;
    qc1 += mm * wq1;
    float wk0 = wk[din * DH_ + d0], wk1 = wk[din * DH_ + d1];
    float wv0 = wv[din * DH_ + d0], wv1 = wv[din * DH_ + d1];
#pragma unroll
    for (int v = 0; v < V_; ++v) {
      float tv = tb[h][v][din];
      kc[v][0] += tv * wk0;
      kc[v][1] += tv * wk1;
      vc[v][0] += tv * wv0;
      vc[v][1] += tv * wv1;
    }
  }
  float s2[V_];
#pragma unroll
  for (int v = 0; v < V_; ++v) {
    float ps = qc0 * kc[v][0] + qc1 * kc[v][1];
    s2[v] = wred_sum(ps) * 0.125f;
  }
  s2[0] += logf(p0 + 1e-6f);
  s2[1] += logf(p1 + 1e-6f);
  s2[2] += logf(p2 + 1e-6f);
  float mx = fmaxf(s2[0], fmaxf(s2[1], s2[2]));
  float e0 = expf(s2[0] - mx), e1 = expf(s2[1] - mx), e2 = expf(s2[2] - mx);
  float inv = 1.f / (e0 + e1 + e2);
  float a0 = e0 * inv, a1 = e1 * inv, a2 = e2 * inv;
  float cr0 = a0 * vc[0][0] + a1 * vc[1][0] + a2 * vc[2][0];
  float cr1 = a0 * vc[0][1] + a1 * vc[1][1] + a2 * vc[2][1];
  cbuf[h][d0] = cr0;
  cbuf[h][d1] = cr1;
  __syncthreads();

  const float* wo = cWo + (size_t)h * DH_ * DH_;
  float co0 = 0.f, co1 = 0.f;
#pragma unroll 8
  for (int din = 0; din < DH_; ++din) {
    float x = cbuf[h][din];
    co0 += x * wo[din * DH_ + d0];
    co1 += x * wo[din * DH_ + d1];
  }
  float gate = gateF[b * 8 + h], beta = betaF[b * 8 + h];
  size_t ob = ((size_t)b * H_ + h) * DH_;
  out[ob + d0] = gate * (beta * co0 + (1.f - beta) * m0);
  out[ob + d1] = gate * (beta * co1 + (1.f - beta) * m1);
}

// ---------------------------------------------------------------------------
// host launcher
// ---------------------------------------------------------------------------
extern "C" void kernel_launch(void* const* d_in, const int* in_sizes, int n_in,
                              void* d_out, int out_size, void* d_ws,
                              size_t ws_size, hipStream_t stream) {
  (void)in_sizes; (void)n_in; (void)out_size; (void)ws_size;
  const float* x_anchor = (const float*)d_in[0];
  const float* x_neighbors = (const float*)d_in[1];
  const float* edge_weights = (const float*)d_in[2];
  const float* g = (const float*)d_in[3];
  const float* aW = (const float*)d_in[4];
  const float* ab = (const float*)d_in[5];
  const float* ag = (const float*)d_in[6];
  const float* abeta = (const float*)d_in[7];
  const float* nW = (const float*)d_in[8];
  const float* nb = (const float*)d_in[9];
  const float* ng = (const float*)d_in[10];
  const float* nbeta = (const float*)d_in[11];
  const float* Wq = (const float*)d_in[12];
  const float* Wk = (const float*)d_in[13];
  const float* Wv = (const float*)d_in[14];
  const float* Wo = (const float*)d_in[15];
  const float* gls = (const float*)d_in[16];
  const float* cWq = (const float*)d_in[17];
  const float* cWk = (const float*)d_in[18];
  const float* cWv = (const float*)d_in[19];
  const float* cWo = (const float*)d_in[20];
  const float* rW1 = (const float*)d_in[21];
  const float* rb1 = (const float*)d_in[22];
  const float* rW2 = (const float*)d_in[23];
  const float* rb2 = (const float*)d_in[24];
  const float* vW = (const float*)d_in[25];
  const float* vb = (const float*)d_in[26];
  const float* mW = (const float*)d_in[27];
  const float* mb = (const float*)d_in[28];
  const float* gW = (const float*)d_in[29];
  const float* gb = (const float*)d_in[30];

  char* ws = (char*)d_ws;
  size_t off = 0;
  auto alloc = [&](size_t bytes) {
    size_t o = off;
    off += (bytes + 255) & ~(size_t)255;
    return o;
  };
  h16* aWpk = (h16*)(ws + alloc((size_t)V_ * DIN_ * REP_ * 2));
  h16* nWpk = (h16*)(ws + alloc((size_t)V_ * DIN_ * REP_ * 2));
  h16* Wqpk = (h16*)(ws + alloc((size_t)V_ * H_ * DH_ * DH_ * 2));
  h16* Wkpk = (h16*)(ws + alloc((size_t)V_ * H_ * DH_ * DH_ * 2));
  h16* Wvpk = (h16*)(ws + alloc((size_t)V_ * H_ * DH_ * DH_ * 2));
  float* haF = (float*)(ws + alloc((size_t)B_ * V_ * REP_ * 4));
  h16* haH = (h16*)(ws + alloc((size_t)B_ * V_ * REP_ * 2));
  float* qF = (float*)(ws + alloc((size_t)B_ * V_ * H_ * DH_ * 4));
  h16* hnH = (h16*)(ws + alloc((size_t)B_ * V_ * K_ * REP_ * 2));
  float* vtF = (float*)(ws + alloc((size_t)B_ * H_ * V_ * DH_ * 4));
  float* piF = (float*)(ws + alloc((size_t)B_ * H_ * V_ * 4));
  float* betaF = (float*)(ws + alloc((size_t)B_ * H_ * 4));
  float* gateF = (float*)(ws + alloc((size_t)B_ * H_ * 4));

  // pack weights to CDNA5 WMMA B-fragment layout (f16)
  k_pack<<<1024, 256, 0, stream>>>(aW, aWpk, DIN_, REP_, V_);
  k_pack<<<1024, 256, 0, stream>>>(nW, nWpk, DIN_, REP_, V_);
  k_pack<<<512, 256, 0, stream>>>(Wq, Wqpk, DH_, DH_, V_ * H_);
  k_pack<<<512, 256, 0, stream>>>(Wk, Wkpk, DH_, DH_, V_ * H_);
  k_pack<<<512, 256, 0, stream>>>(Wv, Wvpk, DH_, DH_, V_ * H_);

  // encoders (WMMA GEMM + register-resident LN), 32 rows per workgroup
  k_enc<<<dim3(B_ / 32, V_), 256, 0, stream>>>(x_anchor, aWpk, ab, ag, abeta,
                                               haF, haH, 0);
  k_enc<<<dim3(B_ * K_ / 32, V_), 256, 0, stream>>>(x_neighbors, nWpk, nb, ng,
                                                    nbeta, nullptr, hnH, 1);
  // q projection (WMMA), 32 rows per workgroup
  k_qgemm<<<dim3(B_ / 32, V_ * H_), 128, 0, stream>>>(haH, Wqpk, qF);
  // fused per-(b,v) attention (WMMA k/v projections)
  k_attn<<<dim3(B_, V_), 256, 0, stream>>>(hnH, qF, Wkpk, Wvpk, Wo, gls,
                                           edge_weights, haF, vtF);
  // router + final cross-view mixing
  k_router<<<B_, 256, 0, stream>>>(g, rW1, rb1, rW2, rb2, vW, vb, mW, mb, gW,
                                   gb, piF, betaF, gateF);
  k_final<<<B_, 256, 0, stream>>>(vtF, piF, betaF, gateF, cWq, cWk, cWv, cWo,
                                  (float*)d_out);
}
